// EquivariantDiffuserV46_66949950210595
// MI455X (gfx1250) — matrix operand order repaired and checked by reading, not compiled, not run on hardware
//
#include <hip/hip_runtime.h>
#include <math.h>

typedef __attribute__((ext_vector_type(16))) _Float16 v16h;
typedef __attribute__((ext_vector_type(8)))  _Float16 v8h;
typedef __attribute__((ext_vector_type(8)))  float    v8f;

#define ND    64    // node feature dim (63 cond + timestep)
#define ED    32    // edge_attr dim
#define HDIM  128   // coord_mlp hidden dim
#define CHUNK 128   // edges per block iteration (8 WMMA N-tiles of 16 edges)
#define HRS   72    // LDS row stride (halves) for gathered h rows (36 dwords)
#define EARS  40    // LDS row stride (halves) for edge_attr rows (20 dwords)
#define HIDRS 40    // LDS row stride (halves) for edge-MLP silu-hidden rows
#define SPRS  (CHUNK + 4)  // padded partial-row stride (dwords)

// Build fp16 node-feature table h = [cond | t]  (BN x 64), and init out = x.
__global__ void prep_kernel(const float* __restrict__ cond, const int* __restrict__ t,
                            const float* __restrict__ x, _Float16* __restrict__ h16,
                            float* __restrict__ out, int BN, int outN) {
  int i = blockIdx.x * blockDim.x + threadIdx.x;
  int total = BN * ND;
  if (i < total) {
    int n = i >> 6, c = i & 63;
    float v = (c < 63) ? cond[n * 63 + c] : (float)(*t);
    h16[i] = (_Float16)v;
  }
  if (i < outN) out[i] = x[i];
}

__device__ __forceinline__ float silu_f(float v) {
  // v * sigmoid(v) with v_exp + v_rcp (TRANS ops co-execute with XDL WMMA)
  return v * __builtin_amdgcn_rcpf(1.0f + __expf(-v));
}

__global__ void __launch_bounds__(256)
egnn_coord_kernel(const float* __restrict__ x,
                  const float* __restrict__ edge_dist,
                  const float* __restrict__ ew1, const float* __restrict__ eb1,
                  const float* __restrict__ ew2, const float* __restrict__ eb2,
                  const float* __restrict__ cw1, const float* __restrict__ cb1,
                  const float* __restrict__ cw2,
                  const int* __restrict__ ei,          // [2][E]
                  const _Float16* __restrict__ h16,    // [BN][64]
                  float* __restrict__ out,             // [BN][3], pre-initialized to x
                  int E, int nChunks) {
  __shared__ __align__(16) _Float16 s_h[2 * CHUNK * HRS];   // [sel][edge][HRS]
  __shared__ __align__(16) _Float16 s_ea[CHUNK * EARS];     // [edge][EARS] edge_attr
  __shared__ __align__(16) _Float16 s_hid[CHUNK * HIDRS];   // [edge][32] silu-hidden
  __shared__ float s_ux[CHUNK], s_uy[CHUNK], s_uz[CHUNK];
  __shared__ int   s_dst[CHUNK];
  __shared__ float s_part[16 * SPRS];                       // [(wv,g)][edge] partials

  const int tid  = threadIdx.x;
  const int lane = tid & 31;
  const int wv   = tid >> 5;    // wave id == hidden M-tile (8 waves x 16 = 128 channels)
  const int g    = lane >> 4;   // lane group (wave32)
  const int lme  = lane & 15;

  // ---- Per-wave resident A operand: 16x160 tile of cw1^T (channels x K). ----
  // A 16x32 in contiguous k order (B matches): lane row M = lme, elem i -> K=16g+i.
  v16h A0, A1, A2, A3, A4;
  v16h E2A0, E2A1;              // ew2^T tiles (out-channels 0..15 / 16..31 x K=32)
  {
    const int ch = 16 * wv + lme;
    #pragma unroll
    for (int i = 0; i < 16; ++i) {
      int kk = 16 * g + i;
      A0[i] = (_Float16)cw1[(kk      ) * HDIM + ch];
      A1[i] = (_Float16)cw1[(kk +  32) * HDIM + ch];
      A2[i] = (_Float16)cw1[(kk +  64) * HDIM + ch];
      A3[i] = (_Float16)cw1[(kk +  96) * HDIM + ch];
      A4[i] = (_Float16)cw1[(kk + 128) * HDIM + ch];
      E2A0[i] = (_Float16)ew2[kk * ED + lme];
      E2A1[i] = (_Float16)ew2[kk * ED + 16 + lme];
    }
  }
  // Bias folded into the WMMA C input. C/D layout: N(edge)=lme, M=8*g+r.
  v8f cinitB, cinitE0, cinitE1;
  float cw2r[8];
  #pragma unroll
  for (int r = 0; r < 8; ++r) {
    cinitB[r]  = cb1[16 * wv + 8 * g + r];
    cw2r[r]    = cw2[16 * wv + 8 * g + r];
    cinitE0[r] = eb2[8 * g + r];
    cinitE1[r] = eb2[16 + 8 * g + r];
  }

  union U { v16h v; v8h h[2]; };

  for (int chunk = blockIdx.x; chunk < nChunks; chunk += gridDim.x) {
    __syncthreads();  // protect LDS reuse across iterations

    // ---------------- Phase A ----------------
    // A1: async gather h[src]/h[dst] rows straight into LDS (CDNA5 async path).
    {
      int eL = tid >> 1, sel = tid & 1;
      int e = chunk * CHUNK + eL;
      int node = (e < E) ? ei[sel * E + e] : 0;
      unsigned long long gp = (unsigned long long)(h16 + (size_t)node * ND);
      unsigned lds = (unsigned)(unsigned long long)(s_h + (sel * CHUNK + eL) * HRS);
      #pragma unroll
      for (int j = 0; j < ND / 8; ++j) {
        asm volatile("global_load_async_to_lds_b128 %0, %1, off offset:%2"
                     :: "v"(lds), "v"(gp), "n"(j * 16) : "memory");
      }
    }
    // A2: per-edge geometry (unit direction, dst index), threads 0..127
    if (tid < CHUNK) {
      int eL = tid;
      int e = chunk * CHUNK + eL;
      float ux = 0.f, uy = 0.f, uz = 0.f; int d = 0;
      if (e < E) {
        int s = ei[e]; d = ei[E + e];
        float dx = x[3 * s + 0] - x[3 * d + 0];
        float dy = x[3 * s + 1] - x[3 * d + 1];
        float dz = x[3 * s + 2] - x[3 * d + 2];
        float len = fmaxf(sqrtf(dx * dx + dy * dy + dz * dz), 1e-8f);
        float inv = 1.0f / len;
        ux = dx * inv; uy = dy * inv; uz = dz * inv;
      }
      s_ux[eL] = ux; s_uy[eL] = uy; s_uz[eL] = uz; s_dst[eL] = d;
    }
    // A3a: edge-MLP layer 1: silu(d*ew1+eb1) -> f16 rows in LDS (B layout).
    // Weight indices wave-uniform -> SMEM path; each k computed exactly once.
    {
      int eL = tid & (CHUNK - 1);
      int hf = __builtin_amdgcn_readfirstlane(tid >> 7);
      int e3 = chunk * CHUNK + eL;
      float dval = (e3 < E) ? edge_dist[e3] : 0.0f;
      const float* w1 = ew1 + hf * 16;
      const float* b1 = eb1 + hf * 16;
      v8h p0, p1;
      #pragma unroll
      for (int i = 0; i < 8; ++i) {
        p0[i] = (_Float16)silu_f(fmaf(dval, w1[i],     b1[i]));
        p1[i] = (_Float16)silu_f(fmaf(dval, w1[8 + i], b1[8 + i]));
      }
      v8h* dstp = (v8h*)(s_hid + eL * HIDRS + hf * 16);
      dstp[0] = p0; dstp[1] = p1;
    }
    __syncthreads();

    // A3b: edge-MLP layer 2 on WMMA: s_ea = eb2 + ew2^T x hid^T.
    // Wave wv handles its own 16 edges; 2 M-tiles (32 out channels).
    {
      const _Float16* hb = s_hid + (wv * 16 + lme) * HIDRS;
      U bh;
      bh.h[0] = *(const v8h*)(hb + 16 * g);
      bh.h[1] = *(const v8h*)(hb + 16 * g + 8);
      v8f c0 = cinitE0, c1 = cinitE1;
      c0 = __builtin_amdgcn_wmma_f32_16x16x32_f16(false, E2A0, false, bh.v, (short)0, c0, false, false);
      c1 = __builtin_amdgcn_wmma_f32_16x16x32_f16(false, E2A1, false, bh.v, (short)0, c1, false, false);
      v8h q0, q1;
      #pragma unroll
      for (int r = 0; r < 8; ++r) { q0[r] = (_Float16)c0[r]; q1[r] = (_Float16)c1[r]; }
      _Float16* eaRow = s_ea + (wv * 16 + lme) * EARS;
      *(v8h*)(eaRow +  0 + 8 * g) = q0;   // channels  0..15: c = 8g+r
      *(v8h*)(eaRow + 16 + 8 * g) = q1;   // channels 16..31
    }
    asm volatile("s_wait_asynccnt 0x0" ::: "memory");
    __syncthreads();

    // ---------------- Phase B: main WMMA (hid^T = cw1^T x m_input^T) ----------
    #pragma unroll
    for (int mt = 0; mt < 8; ++mt) {
      const int er = mt * 16 + lme;     // edge handled by this lane (N column)
      const _Float16* ms = s_h + (0 * CHUNK + er) * HRS;
      const _Float16* md = s_h + (1 * CHUNK + er) * HRS;
      const _Float16* me = s_ea + er * EARS;
      U b0, b1, b2, b3, b4;
      b0.h[0] = *(const v8h*)(ms + 16 * g);      b0.h[1] = *(const v8h*)(ms + 16 * g + 8);
      b1.h[0] = *(const v8h*)(ms + 32 + 16 * g); b1.h[1] = *(const v8h*)(ms + 32 + 16 * g + 8);
      b2.h[0] = *(const v8h*)(md + 16 * g);      b2.h[1] = *(const v8h*)(md + 16 * g + 8);
      b3.h[0] = *(const v8h*)(md + 32 + 16 * g); b3.h[1] = *(const v8h*)(md + 32 + 16 * g + 8);
      b4.h[0] = *(const v8h*)(me + 16 * g);      b4.h[1] = *(const v8h*)(me + 16 * g + 8);

      v8f c = cinitB;   // bias pre-loaded into the accumulator
      c = __builtin_amdgcn_wmma_f32_16x16x32_f16(false, A0, false, b0.v, (short)0, c, false, false);
      c = __builtin_amdgcn_wmma_f32_16x16x32_f16(false, A1, false, b1.v, (short)0, c, false, false);
      c = __builtin_amdgcn_wmma_f32_16x16x32_f16(false, A2, false, b2.v, (short)0, c, false, false);
      c = __builtin_amdgcn_wmma_f32_16x16x32_f16(false, A3, false, b3.v, (short)0, c, false, false);
      c = __builtin_amdgcn_wmma_f32_16x16x32_f16(false, A4, false, b4.v, (short)0, c, false, false);

      // Channel (M) reduction: per-lane over the 8 accumulator VGPRs.
      float acc = 0.0f;
      #pragma unroll
      for (int r = 0; r < 8; ++r)
        acc = fmaf(silu_f(c[r]), cw2r[r], acc);
      s_part[(wv * 2 + g) * SPRS + mt * 16 + lme] = acc;
    }
    __syncthreads();

    // ---------------- Phase C: combine + scatter ----------------
    if (tid < CHUNK) {
      int e = chunk * CHUNK + tid;
      if (e < E) {
        float cwght = 0.0f;
        #pragma unroll
        for (int p = 0; p < 16; ++p) cwght += s_part[p * SPRS + tid];
        int d = s_dst[tid];
        atomicAdd(&out[3 * d + 0], cwght * s_ux[tid]);
        atomicAdd(&out[3 * d + 1], cwght * s_uy[tid]);
        atomicAdd(&out[3 * d + 2], cwght * s_uz[tid]);
      }
    }
  }
}

extern "C" void kernel_launch(void* const* d_in, const int* in_sizes, int n_in,
                              void* d_out, int out_size, void* d_ws, size_t ws_size,
                              hipStream_t stream) {
  (void)n_in; (void)ws_size;
  const float* x     = (const float*)d_in[0];
  const float* cond  = (const float*)d_in[1];
  const float* edist = (const float*)d_in[2];
  const float* ew1   = (const float*)d_in[3];
  const float* eb1   = (const float*)d_in[4];
  const float* ew2   = (const float*)d_in[5];
  const float* eb2   = (const float*)d_in[6];
  // d_in[7..10] = node_mlp weights: dead w.r.t. returned coordinates -> skipped.
  const float* cw1   = (const float*)d_in[11];
  const float* cb1   = (const float*)d_in[12];
  const float* cw2   = (const float*)d_in[13];
  const int*   ei    = (const int*)d_in[14];
  const int*   t     = (const int*)d_in[15];

  const int BN = in_sizes[0] / 3;     // 50000
  const int E  = in_sizes[2];         // 800000
  float* out = (float*)d_out;
  _Float16* h16 = (_Float16*)d_ws;    // BN*64*2 = 6.4 MB of workspace

  int prepThreads = BN * ND;          // > out_size, covers out-init too
  int pb = (prepThreads + 255) / 256;
  prep_kernel<<<pb, 256, 0, stream>>>(cond, t, x, h16, out, BN, out_size);

  int nChunks = (E + CHUNK - 1) / CHUNK;   // 6250
  int blocks = nChunks < 1024 ? nChunks : 1024;
  egnn_coord_kernel<<<blocks, 256, 0, stream>>>(x, edist, ew1, eb1, ew2, eb2,
                                                cw1, cb1, cw2, ei, h16, out,
                                                E, nChunks);
}
